// SpatialSelfCorrelation_82291573391858
// MI455X (gfx1250) — matrix operand order repaired and checked by reading, not compiled, not run on hardware
//
#include <hip/hip_runtime.h>

// ---------------------------------------------------------------------------
// SpatialSelfCorrelation on MI455X (gfx1250, wave32, WMMA)
//   y = Wo @ ( xf @ softmax( (W1@xf+b1) @ (W2@xf+b2)^T ) ) + bo
// Five batched 1024^3 GEMMs (B=64) -> compute-bound -> bf16 WMMA with f32
// accumulation. Staging uses GLOBAL_LOAD_ASYNC_TO_LDS_B128 (ASYNCcnt) where
// available; NN B-operand transpose packs via v_perm_b32.
// ---------------------------------------------------------------------------

typedef __attribute__((ext_vector_type(16))) __bf16 v16bf;
typedef __attribute__((ext_vector_type(8)))  float  v8f;

#define LDIM 1024      // all matrices are 1024x1024, ld = 1024
#define BM   128
#define BN   128
#define BK   32
#define PAD  8
#define SROW (BK + PAD)   // 40 bf16 per LDS row (80B) -> conflict-free b128 reads

#if defined(__has_builtin)
#if __has_builtin(__builtin_amdgcn_global_load_async_to_lds_b128)
#define HAVE_ASYNC_LDS 1
#endif
#endif

#ifdef HAVE_ASYNC_LDS
// builtin pointee type is a GCC-vector int4 (per clang diagnostic)
typedef int v4i __attribute__((vector_size(16)));
typedef __attribute__((address_space(1))) v4i glob_v4i;
typedef __attribute__((address_space(3))) v4i lds_v4i;
__device__ __forceinline__ void async_copy_b128(const void* g, void* l) {
  // generic->AS via integer cast: LDS generic ptr low 32 bits == LDS offset;
  // global generic ptr bit pattern == AS1 address. int->ptr casts are always
  // legal and avoid addrspacecast legality questions.
  __builtin_amdgcn_global_load_async_to_lds_b128(
      (glob_v4i*)(unsigned long long)g,
      (lds_v4i*)(unsigned int)(unsigned long long)l, 0, 0);
}
__device__ __forceinline__ void wait_async() {
#if __has_builtin(__builtin_amdgcn_s_wait_asynccnt)
  __builtin_amdgcn_s_wait_asynccnt(0);
#else
  asm volatile("s_wait_asynccnt 0x0" ::: "memory");
#endif
}
#endif

union FragAB {
  v16bf v;
  uint4 u4[2];
};

__device__ __forceinline__ unsigned short f2bf(float f) {
  unsigned u = __float_as_uint(f);
  u += 0x7FFFu + ((u >> 16) & 1u);   // round-to-nearest-even
  return (unsigned short)(u >> 16);
}

// ---------------------------------------------------------------------------
// f32 -> bf16 conversion (vectorized: 4 floats in, 4 bf16 out)
// ---------------------------------------------------------------------------
__global__ __launch_bounds__(256) void cvt_f32_to_bf16(
    const float* __restrict__ src, unsigned short* __restrict__ dst, int n4) {
  int i = blockIdx.x * blockDim.x + threadIdx.x;
  if (i < n4) {
    const float4 f = ((const float4*)src)[i];
    uint2 o;
    o.x = (unsigned)f2bf(f.x) | ((unsigned)f2bf(f.y) << 16);
    o.y = (unsigned)f2bf(f.z) | ((unsigned)f2bf(f.w) << 16);
    ((uint2*)dst)[i] = o;
  }
}

// ---------------------------------------------------------------------------
// Batched 1024x1024x1024 bf16 GEMM, f32 accumulate.
//   D[bz] = A[bz] (MxK, row-major) * op(B[bz]) + bias
//   TRANSB=0: B is KxN row-major (NN)   -> transpose while staging to LDS
//   TRANSB=1: B is NxK row-major (NT)   -> straight async copy to LDS
// LDS: As row-major MxK; Bs holds B^T (N rows, K contiguous) so A and B WMMA
// fragments are each two ds_load_b128 matching the CDNA5 VGPR layouts.
// ---------------------------------------------------------------------------
template <bool TRANSB, bool OUTBF16>
__global__ __launch_bounds__(256) void gemm1024_bf16(
    const unsigned short* __restrict__ A, long long strideA,
    const unsigned short* __restrict__ B, long long strideB,
    void* __restrict__ D, long long strideD, const float* __restrict__ bias) {
  __shared__ __align__(16) unsigned short As[2][BM * SROW];
  __shared__ __align__(16) unsigned short Bs[2][BN * SROW];

  const int tid   = threadIdx.x;
  const int lane  = tid & 31;
  const int wave  = tid >> 5;
  const int waveM = wave >> 2;   // 0..1  -> 64 rows each
  const int waveN = wave & 3;    // 0..3  -> 32 cols each
  const int half  = lane >> 4;   // K-half select per ISA layout
  const int l16   = lane & 15;

  const int n0 = blockIdx.x * BN;
  const int m0 = blockIdx.y * BM;
  const int bz = blockIdx.z;

  const unsigned short* gA = A + (long long)bz * strideA;
  const unsigned short* gB = B + (long long)bz * strideB;

  // straight 128x32 tile copy: 512 uint4, thread t handles #t and #(t+256)
  const int ar0 = tid >> 2, ac0 = (tid & 3) * 8;
  const int ar1 = ar0 + 64, ac1 = ac0;
  // NN B tile (32K x 128N): thread handles a 2(K) x 8(N) block
  const int bkk = (tid >> 4) * 2;   // even K row
  const int bnn = (tid & 15) * 8;   // N offset

  uint4 ra0, ra1, rb0, rb1;
  (void)ra0; (void)ra1;

  // -- issue phase: start global traffic for tile kt into buffer `buf` ------
  auto stage_issue = [&](int buf, int kt) {
    const unsigned short* pa = gA + (long long)m0 * LDIM + kt * BK;
#ifdef HAVE_ASYNC_LDS
    async_copy_b128(pa + (long long)ar0 * LDIM + ac0, &As[buf][ar0 * SROW + ac0]);
    async_copy_b128(pa + (long long)ar1 * LDIM + ac1, &As[buf][ar1 * SROW + ac1]);
    if (TRANSB) {
      const unsigned short* pb = gB + (long long)n0 * LDIM + kt * BK;
      async_copy_b128(pb + (long long)ar0 * LDIM + ac0, &Bs[buf][ar0 * SROW + ac0]);
      async_copy_b128(pb + (long long)ar1 * LDIM + ac1, &Bs[buf][ar1 * SROW + ac1]);
    }
#else
    ra0 = *(const uint4*)(pa + (long long)ar0 * LDIM + ac0);
    ra1 = *(const uint4*)(pa + (long long)ar1 * LDIM + ac1);
    if (TRANSB) {
      const unsigned short* pb = gB + (long long)n0 * LDIM + kt * BK;
      rb0 = *(const uint4*)(pb + (long long)ar0 * LDIM + ac0);
      rb1 = *(const uint4*)(pb + (long long)ar1 * LDIM + ac1);
    }
#endif
    if (!TRANSB) {  // B row-major KxN: 2 K-rows x 8 N-values into regs
      const unsigned short* pb =
          gB + (long long)(kt * BK + bkk) * LDIM + n0 + bnn;
      rb0 = *(const uint4*)(pb);
      rb1 = *(const uint4*)(pb + LDIM);
      if (kt + 1 < LDIM / BK)  // L2 warm-up -> global_prefetch_b8
        __builtin_prefetch(pb + BK * LDIM, 0, 0);
    }
  };
  // -- commit phase: VGPR->LDS stores (only paths that can't go async) ------
  auto stage_commit = [&](int buf) {
#ifndef HAVE_ASYNC_LDS
    *(uint4*)&As[buf][ar0 * SROW + ac0] = ra0;
    *(uint4*)&As[buf][ar1 * SROW + ac1] = ra1;
    if (TRANSB) {
      *(uint4*)&Bs[buf][ar0 * SROW + ac0] = rb0;
      *(uint4*)&Bs[buf][ar1 * SROW + ac1] = rb1;
    }
#endif
    if (!TRANSB) {
      const unsigned* w0 = (const unsigned*)&rb0;  // row k   (8 bf16)
      const unsigned* w1 = (const unsigned*)&rb1;  // row k+1 (8 bf16)
#pragma unroll
      for (int i = 0; i < 8; ++i) {
        // {B[k][n+i], B[k+1][n+i]} packed with one v_perm_b32
        unsigned v = __builtin_amdgcn_perm(
            w1[i >> 1], w0[i >> 1], (i & 1) ? 0x07060302u : 0x05040100u);
        *(unsigned*)&Bs[buf][(bnn + i) * SROW + bkk] = v;
      }
    }
  };

  v8f acc[4][2];
#pragma unroll
  for (int mf = 0; mf < 4; ++mf)
#pragma unroll
    for (int nf = 0; nf < 2; ++nf)
      acc[mf][nf] = (v8f){0.f, 0.f, 0.f, 0.f, 0.f, 0.f, 0.f, 0.f};

  const int KT = LDIM / BK;   // 32 K-steps

  stage_issue(0, 0);
  stage_commit(0);
#ifdef HAVE_ASYNC_LDS
  wait_async();
#endif
  __syncthreads();

  for (int kt = 0; kt < KT; ++kt) {
    const int cur = kt & 1;
    if (kt + 1 < KT) stage_issue(cur ^ 1, kt + 1);  // overlap with WMMA below

    FragAB afrag[4], bfrag[2];
#pragma unroll
    for (int mf = 0; mf < 4; ++mf) {
      // A 16x32 bf16 layout: lanes 0-15 K{0..7,16..23}, lanes 16-31 +8
      const unsigned short* r = &As[cur][(waveM * 64 + mf * 16 + l16) * SROW];
      afrag[mf].u4[0] = *(const uint4*)(r + 8 * half);
      afrag[mf].u4[1] = *(const uint4*)(r + 16 + 8 * half);
    }
#pragma unroll
    for (int nf = 0; nf < 2; ++nf) {
      // B 32x16 layout: lanes 0-15 K{0..7,16..23}, lanes 16-31 +8; Bs is N-major
      const unsigned short* r = &Bs[cur][(waveN * 32 + nf * 16 + l16) * SROW];
      bfrag[nf].u4[0] = *(const uint4*)(r + 8 * half);
      bfrag[nf].u4[1] = *(const uint4*)(r + 16 + 8 * half);
    }
#pragma unroll
    for (int mf = 0; mf < 4; ++mf)
#pragma unroll
      for (int nf = 0; nf < 2; ++nf)
        acc[mf][nf] = __builtin_amdgcn_wmma_f32_16x16x32_bf16(
            false, afrag[mf].v, false, bfrag[nf].v, (short)0, acc[mf][nf],
            false, false);

    if (kt + 1 < KT) {
      stage_commit(cur ^ 1);
#ifdef HAVE_ASYNC_LDS
      wait_async();       // async copies for kt+1 done (overlapped the WMMAs)
#endif
    }
    __syncthreads();
  }

  // epilogue: C/D layout -> VGPR r, lane<16: (M=r,N=lane); lane>=16: (M=8+r)
#pragma unroll
  for (int mf = 0; mf < 4; ++mf) {
    const int mb = m0 + waveM * 64 + mf * 16 + 8 * half;
#pragma unroll
    for (int nf = 0; nf < 2; ++nf) {
      const int n = n0 + waveN * 32 + nf * 16 + l16;
#pragma unroll
      for (int r = 0; r < 8; ++r) {
        const int m = mb + r;
        float v = acc[mf][nf][r];
        if (bias) v += bias[m];
        const long long idx = (long long)bz * strideD + (long long)m * LDIM + n;
        if (OUTBF16)
          ((unsigned short*)D)[idx] = f2bf(v);
        else
          ((float*)D)[idx] = v;
      }
    }
  }
}

// ---------------------------------------------------------------------------
// Row softmax: one block (256 threads, 8 waves) per 1024-wide row, f32 in,
// bf16 probabilities out. wave32 shuffle reductions + cross-wave LDS.
// ---------------------------------------------------------------------------
__global__ __launch_bounds__(256) void softmax_rows(
    const float* __restrict__ S, unsigned short* __restrict__ P) {
  __shared__ float red[8];
  const long long row = blockIdx.x;
  const float* s = S + row * LDIM;
  unsigned short* p = P + row * LDIM;
  const int tid = threadIdx.x;

  float v[4];
  float m = -3.0e38f;
#pragma unroll
  for (int i = 0; i < 4; ++i) {
    v[i] = s[tid + i * 256];
    m = fmaxf(m, v[i]);
  }
#pragma unroll
  for (int off = 16; off > 0; off >>= 1) m = fmaxf(m, __shfl_xor(m, off, 32));
  if ((tid & 31) == 0) red[tid >> 5] = m;
  __syncthreads();
  m = red[0];
#pragma unroll
  for (int i = 1; i < 8; ++i) m = fmaxf(m, red[i]);
  __syncthreads();

  float sum = 0.f;
#pragma unroll
  for (int i = 0; i < 4; ++i) {
    v[i] = __expf(v[i] - m);
    sum += v[i];
  }
#pragma unroll
  for (int off = 16; off > 0; off >>= 1) sum += __shfl_xor(sum, off, 32);
  if ((tid & 31) == 0) red[tid >> 5] = sum;
  __syncthreads();
  sum = 0.f;
#pragma unroll
  for (int i = 0; i < 8; ++i) sum += red[i];
  const float inv = 1.0f / sum;
#pragma unroll
  for (int i = 0; i < 4; ++i) p[tid + i * 256] = f2bf(v[i] * inv);
}

// ---------------------------------------------------------------------------
// Orchestration. Workspace layout (bytes):
//   xh  @ 0          : 128 MB  bf16 x            (64*1024*1024)
//   qh  @ 128 MB     : 128 MB  bf16 q, reused for P = softmax(S)
//   kh  @ 256 MB     : 128 MB  bf16 k, reused for out = xf@P
//   S   @ 384 MB     : 256 MB  f32 attention logits
//   w1h/w2h/woh      : 3 x 2 MB bf16 weights
// ---------------------------------------------------------------------------
extern "C" void kernel_launch(void* const* d_in, const int* in_sizes, int n_in,
                              void* d_out, int out_size, void* d_ws,
                              size_t ws_size, hipStream_t stream) {
  (void)in_sizes; (void)n_in; (void)out_size; (void)ws_size;
  const float* x  = (const float*)d_in[0];
  const float* W1 = (const float*)d_in[1];
  const float* b1 = (const float*)d_in[2];
  const float* W2 = (const float*)d_in[3];
  const float* b2 = (const float*)d_in[4];
  const float* Wo = (const float*)d_in[5];
  const float* bo = (const float*)d_in[6];
  float* y = (float*)d_out;

  char* ws = (char*)d_ws;
  const long long BS = 1024LL * 1024LL;            // per-batch matrix elems
  unsigned short* xh  = (unsigned short*)(ws);
  unsigned short* qh  = (unsigned short*)(ws + 134217728LL);
  unsigned short* kh  = (unsigned short*)(ws + 268435456LL);
  float*          Sb  = (float*)(ws + 402653184LL);
  unsigned short* w1h = (unsigned short*)(ws + 671088640LL);
  unsigned short* w2h = (unsigned short*)(ws + 673185792LL);
  unsigned short* woh = (unsigned short*)(ws + 675282944LL);

  // f32 -> bf16 conversions
  cvt_f32_to_bf16<<<65536, 256, 0, stream>>>(x, xh, 64 * 1024 * 1024 / 4);
  cvt_f32_to_bf16<<<1024, 256, 0, stream>>>(W1, w1h, 1024 * 1024 / 4);
  cvt_f32_to_bf16<<<1024, 256, 0, stream>>>(W2, w2h, 1024 * 1024 / 4);
  cvt_f32_to_bf16<<<1024, 256, 0, stream>>>(Wo, woh, 1024 * 1024 / 4);

  dim3 grid(LDIM / BN, LDIM / BM, 64);   // 8 x 8 x 64 = 4096 workgroups

  // q = W1 @ xf + b1   (NN, weights batch-stride 0), bf16 out
  gemm1024_bf16<false, true><<<grid, 256, 0, stream>>>(w1h, 0, xh, BS, qh, BS, b1);
  // k = W2 @ xf + b2
  gemm1024_bf16<false, true><<<grid, 256, 0, stream>>>(w2h, 0, xh, BS, kh, BS, b2);
  // S = q @ k^T  (NT), f32 out
  gemm1024_bf16<true, false><<<grid, 256, 0, stream>>>(qh, BS, kh, BS, Sb, BS, nullptr);
  // P = softmax(S) rows -> bf16, overwrite qh
  softmax_rows<<<64 * 1024, 256, 0, stream>>>(Sb, qh);
  // out = xf @ P  (NN), bf16 out -> overwrite kh
  gemm1024_bf16<false, true><<<grid, 256, 0, stream>>>(xh, BS, qh, BS, kh, BS, nullptr);
  // y = Wo @ out + bo  (NN), f32 out -> d_out
  gemm1024_bf16<false, false><<<grid, 256, 0, stream>>>(woh, 0, kh, BS, y, BS, bo);
}
